// IoUNet_20083267076676
// MI455X (gfx1250) — matrix operand order repaired and checked by reading, not compiled, not run on hardware
//
#include <hip/hip_runtime.h>
#include <hip/hip_bf16.h>
#include <hip/hip_fp16.h>

// ---------------------------------------------------------------------------
// IoU-Net head for MI455X (gfx1250): ROI-align + 2x conv3x3 + FC, all matmuls
// via v_wmma_f32_16x16x32_f16 (wave32). ~102 GFLOP, compute-bound; feat_map
// (64MB) is L2-resident, per-ROI intermediates live in LDS.
// Round 2: register-blocked WMMA (A-fragment reuse across 8/4 N-tiles) to cut
// VMEM/DS issues per WMMA; position-major ROI-align gather for coalescing.
// ---------------------------------------------------------------------------

typedef __attribute__((ext_vector_type(16))) _Float16 v16h;
typedef __attribute__((ext_vector_type(8)))  _Float16 v8h;
typedef __attribute__((ext_vector_type(8)))  float    v8f;

union FragA { v16h v; v8h h[2]; };

#define C_IN   64
#define C_MID  128
#define FC_DIM 256
#define FIN    (C_MID * 49)   // 6272
#define NROI   4096
#define PADW   9

static __device__ __forceinline__ v8f wmma_f16(v16h a, v16h b, v8f c) {
    // 8 args: (neg_a, A, neg_b, B, c_mod, C, reuse_a, reuse_b)
    return __builtin_amdgcn_wmma_f32_16x16x32_f16(false, a, false, b,
                                                  (short)0, c, false, false);
}

// ---------------------------------------------------------------------------
// Kernel 0: pack weights to f16 in WMMA-friendly layouts.
//   W1h[n][k], k = (ky*3+kx)*64  + ci   (n in [0,128), k in [0,576))
//   W2h[n][k], k = (ky*3+kx)*128 + ci   (n in [0,128), k in [0,1152))
//   FW1h[n][f] = fw1[f][n]              (n in [0,256), f in [0,6272))
// ---------------------------------------------------------------------------
__global__ __launch_bounds__(256) void pack_weights(
    const float* __restrict__ w1, const float* __restrict__ w2,
    const float* __restrict__ fw1,
    _Float16* __restrict__ W1h, _Float16* __restrict__ W2h,
    _Float16* __restrict__ FW1h) {
    int tid = blockIdx.x * blockDim.x + threadIdx.x;
    if (tid < 128 * 576) {
        int n = tid / 576, k = tid - n * 576;
        int pos = k >> 6, ci = k & 63;                 // w1: [O][I][3][3]
        W1h[tid] = (_Float16)w1[(n * 64 + ci) * 9 + pos];
    }
    if (tid < 128 * 1152) {
        int n = tid / 1152, k = tid - n * 1152;
        int pos = k >> 7, ci = k & 127;
        W2h[tid] = (_Float16)w2[(n * 128 + ci) * 9 + pos];
    }
    if (tid < 256 * 6272) {
        int n = tid / 6272, f = tid - n * 6272;        // fw1: [6272][256]
        FW1h[tid] = (_Float16)fw1[f * 256 + n];
    }
}

// ---------------------------------------------------------------------------
// Register-blocked implicit-GEMM conv3x3 (SAME) layer: one wave owns m_tile =
// wave (16 output positions) x all 8 n-tiles (128 channels), K = 9*CI.
// A gathered from HWC zero-padded 9x9xCI f16 LDS tile (loaded ONCE per k-step,
// reused by 8 WMMAs); B = [N][K] f16 in global (L0/L2 resident weights).
// Returns accumulators in acc[8].
// ---------------------------------------------------------------------------
template<int CI>
static __device__ __forceinline__ void conv_layer_acc(
    const _Float16* __restrict__ sBuf, const _Float16* __restrict__ W,
    int wave, int lane, v8f acc[8]) {
    const int K = 9 * CI;
    const int SHIFT = (CI == 64) ? 6 : 7;
    const int half = lane >> 4;
    int p  = wave * 16 + (lane & 15);    // m_tile == wave
    int pc = p < 49 ? p : 48;            // clamp for addressing; rows p>=49 discarded
    int oy = pc / 7, ox = pc - oy * 7;
    const _Float16* wbase = W + (lane & 15) * K + half * 8;
    v8f z = {};
    #pragma unroll
    for (int nt = 0; nt < 8; ++nt) acc[nt] = z;
    for (int kk = 0; kk < K; kk += 32) {
        FragA a;
        int kb0 = kk + half * 8;         // lanes 0-15: K 0-7,16-23 ; 16-31: 8-15,24-31
        int kb1 = kb0 + 16;
        int pos0 = kb0 >> SHIFT, ci0 = kb0 & (CI - 1);
        int pos1 = kb1 >> SHIFT, ci1 = kb1 & (CI - 1);
        a.h[0] = *(const v8h*)(sBuf + ((oy + pos0 / 3) * PADW + ox + pos0 % 3) * CI + ci0);
        a.h[1] = *(const v8h*)(sBuf + ((oy + pos1 / 3) * PADW + ox + pos1 % 3) * CI + ci1);
        #pragma unroll
        for (int nt = 0; nt < 8; ++nt) {           // A reused 8x
            FragA b;
            const _Float16* wrow = wbase + nt * 16 * K + kk;
            b.h[0] = *(const v8h*)(wrow);
            b.h[1] = *(const v8h*)(wrow + 16);
            acc[nt] = wmma_f16(a.v, b.v, acc[nt]);
        }
    }
}

// ---------------------------------------------------------------------------
// Kernel 1: fused ROI-align + conv1 + ReLU + conv2 + ReLU per ROI.
// 1 block = 1 ROI, 128 threads = 4 waves (wave = m_tile).
// Output X2[roi][c*49 + p] f16 (= reference's NCHW flatten order).
// ---------------------------------------------------------------------------
__global__ __launch_bounds__(128) void roi_conv_kernel(
    const float* __restrict__ feat, const float* __restrict__ boxes,
    const _Float16* __restrict__ W1h, const float* __restrict__ b1,
    const _Float16* __restrict__ W2h, const float* __restrict__ b2,
    _Float16* __restrict__ X2) {
    __shared__ _Float16 sIn [PADW * PADW * C_IN];    // 10368 B, HWC padded
    __shared__ _Float16 sMid[PADW * PADW * C_MID];   // 20736 B, HWC padded

    const int roi  = blockIdx.x;
    const int tid  = threadIdx.x;
    const int wave = tid >> 5;
    const int lane = tid & 31;

    // zero both tiles (establishes the SAME-padding border)
    unsigned* zi = (unsigned*)sIn;
    for (int i = tid; i < PADW * PADW * C_IN / 2; i += 128)  zi[i] = 0u;
    unsigned* zm = (unsigned*)sMid;
    for (int i = tid; i < PADW * PADW * C_MID / 2; i += 128) zm[i] = 0u;
    __syncthreads();

    // ---- ROI align (pool=7, sr=2, aligned=False) -> sIn interior ----
    // position-major mapping: consecutive lanes take consecutive output
    // positions -> bilinear taps of adjacent lanes share cachelines in L2.
    const float* bx = boxes + roi * 5;
    int   bidx = (int)bx[0];
    float x1 = bx[1], y1 = bx[2], x2c = bx[3], y2c = bx[4];
    float roiw = fmaxf(x2c - x1, 1.0f), roih = fmaxf(y2c - y1, 1.0f);
    float binw = roiw * (1.0f / 7.0f),  binh = roih * (1.0f / 7.0f);
    const float* fbase = feat + (size_t)bidx * C_IN * 256 * 256;

    for (int i = tid; i < 49 * C_IN; i += 128) {
        int c = i / 49, o = i - c * 49;
        int oy = o / 7, ox = o - oy * 7;
        const float* fc = fbase + (size_t)c * 65536;
        float acc = 0.0f;
        #pragma unroll
        for (int s = 0; s < 4; ++s) {
            int sy = 2 * oy + (s >> 1), sx = 2 * ox + (s & 1);
            float yc = y1 + binh * ((sy + 0.5f) * 0.5f);
            float xc = x1 + binw * ((sx + 0.5f) * 0.5f);
            bool v = (yc >= -1.0f) && (yc <= 256.0f) && (xc >= -1.0f) && (xc <= 256.0f);
            float cy = fminf(fmaxf(yc, 0.0f), 255.0f);
            float cx = fminf(fmaxf(xc, 0.0f), 255.0f);
            int y0 = (int)cy, x0 = (int)cx;
            int y1i = min(y0 + 1, 255), x1i = min(x0 + 1, 255);
            float ly = cy - (float)y0, lx = cx - (float)x0;
            float hy = 1.0f - ly, hx = 1.0f - lx;
            float bv = hy * hx * fc[y0  * 256 + x0 ] + hy * lx * fc[y0  * 256 + x1i]
                     + ly * hx * fc[y1i * 256 + x0 ] + ly * lx * fc[y1i * 256 + x1i];
            acc += v ? bv : 0.0f;
        }
        sIn[((oy + 1) * PADW + (ox + 1)) * C_IN + c] = (_Float16)(acc * 0.25f);
    }
    __syncthreads();

    const int half = lane >> 4;

    // ---- conv1: [49x576]x[576x128] WMMA, A reused across 8 n-tiles ----
    {
        v8f acc[8];
        conv_layer_acc<C_IN>(sIn, W1h, wave, lane, acc);
        #pragma unroll
        for (int nt = 0; nt < 8; ++nt) {
            int n = nt * 16 + (lane & 15);
            float bias = b1[n];
            #pragma unroll
            for (int j = 0; j < 8; ++j) {
                int p = wave * 16 + j + 8 * half;      // D: VGPR j -> M = j(+8)
                if (p < 49) {
                    float v = acc[nt][j] + bias;
                    v = v > 0.0f ? v : 0.0f;
                    int oy = p / 7, ox = p - (p / 7) * 7;
                    sMid[((oy + 1) * PADW + (ox + 1)) * C_MID + n] = (_Float16)v;
                }
            }
        }
    }
    __syncthreads();

    // ---- conv2: [49x1152]x[1152x128] WMMA -> X2[roi][n*49+p] ----
    {
        v8f acc[8];
        conv_layer_acc<C_MID>(sMid, W2h, wave, lane, acc);
        _Float16* xrow = X2 + (size_t)roi * FIN;
        #pragma unroll
        for (int nt = 0; nt < 8; ++nt) {
            int n = nt * 16 + (lane & 15);
            float bias = b2[n];
            #pragma unroll
            for (int j = 0; j < 8; ++j) {
                int p = wave * 16 + j + 8 * half;
                if (p < 49) {
                    float v = acc[nt][j] + bias;
                    v = v > 0.0f ? v : 0.0f;
                    xrow[n * 49 + p] = (_Float16)v;
                }
            }
        }
    }
}

// ---------------------------------------------------------------------------
// Kernel 2: FC1 GEMM [4096 x 6272] x [6272 x 256] + ReLU via WMMA.
// 256 blocks x 128 threads: block = one M-tile; wave w owns n-tiles 4w..4w+3,
// A fragment loaded once per k-step and reused by 4 WMMAs.
// ---------------------------------------------------------------------------
__global__ __launch_bounds__(128) void fc1_kernel(
    const _Float16* __restrict__ X2, const _Float16* __restrict__ FW1h,
    const float* __restrict__ fb1, float* __restrict__ FC1) {
    const int lane = threadIdx.x & 31;
    const int wave = threadIdx.x >> 5;
    const int m_tile = blockIdx.x;
    const int half = lane >> 4;
    const int mrow = m_tile * 16 + (lane & 15);
    const _Float16* arow = X2 + (size_t)mrow * FIN + half * 8;
    const _Float16* brow0 = FW1h + (size_t)((wave * 4 + 0) * 16 + (lane & 15)) * FIN + half * 8;
    const _Float16* brow1 = FW1h + (size_t)((wave * 4 + 1) * 16 + (lane & 15)) * FIN + half * 8;
    const _Float16* brow2 = FW1h + (size_t)((wave * 4 + 2) * 16 + (lane & 15)) * FIN + half * 8;
    const _Float16* brow3 = FW1h + (size_t)((wave * 4 + 3) * 16 + (lane & 15)) * FIN + half * 8;
    const _Float16* brows[4] = {brow0, brow1, brow2, brow3};
    v8f z = {};
    v8f acc[4] = {z, z, z, z};
    #pragma unroll 2
    for (int kk = 0; kk < FIN; kk += 32) {
        FragA a;
        a.h[0] = *(const v8h*)(arow + kk);        // K kk+half*8 .. +7
        a.h[1] = *(const v8h*)(arow + kk + 16);   // K kk+16+half*8 .. +7
        #pragma unroll
        for (int nt = 0; nt < 4; ++nt) {          // A reused 4x
            FragA b;
            b.h[0] = *(const v8h*)(brows[nt] + kk);      // K kk+half*8 .. +7
            b.h[1] = *(const v8h*)(brows[nt] + kk + 16);
            acc[nt] = wmma_f16(a.v, b.v, acc[nt]);
        }
    }
    #pragma unroll
    for (int nt = 0; nt < 4; ++nt) {
        int n = (wave * 4 + nt) * 16 + (lane & 15);
        float bias = fb1[n];
        #pragma unroll
        for (int j = 0; j < 8; ++j) {
            int m = m_tile * 16 + j + 8 * half;
            float v = acc[nt][j] + bias;
            FC1[m * FC_DIM + n] = v > 0.0f ? v : 0.0f;
        }
    }
}

// ---------------------------------------------------------------------------
// Kernel 3: FC2 (256 -> 1) + sigmoid. One wave per ROI, shuffle reduction.
// ---------------------------------------------------------------------------
__global__ __launch_bounds__(128) void fc2_kernel(
    const float* __restrict__ FC1, const float* __restrict__ fw2,
    const float* __restrict__ fb2, float* __restrict__ out) {
    const int lane = threadIdx.x & 31;
    const int wave = threadIdx.x >> 5;
    const int roi  = blockIdx.x * 4 + wave;
    const float* row = FC1 + (size_t)roi * FC_DIM;
    float s = 0.0f;
    #pragma unroll
    for (int j = lane; j < FC_DIM; j += 32) s += row[j] * fw2[j];
    #pragma unroll
    for (int off = 16; off > 0; off >>= 1) s += __shfl_down(s, off, 32);
    if (lane == 0) out[roi] = 1.0f / (1.0f + __expf(-(s + fb2[0])));
}

// ---------------------------------------------------------------------------
extern "C" void kernel_launch(void* const* d_in, const int* in_sizes, int n_in,
                              void* d_out, int out_size, void* d_ws, size_t ws_size,
                              hipStream_t stream) {
    const float* feat  = (const float*)d_in[0];
    const float* boxes = (const float*)d_in[1];
    const float* w1    = (const float*)d_in[2];
    const float* b1    = (const float*)d_in[3];
    const float* w2    = (const float*)d_in[4];
    const float* b2    = (const float*)d_in[5];
    const float* fw1   = (const float*)d_in[6];
    const float* fb1   = (const float*)d_in[7];
    const float* fw2   = (const float*)d_in[8];
    const float* fb2   = (const float*)d_in[9];
    float* out = (float*)d_out;

    char* ws = (char*)d_ws;
    size_t off = 0;
    auto alloc = [&](size_t bytes) -> void* {
        void* p = (void*)(ws + off);
        off += (bytes + 255) & ~(size_t)255;
        return p;
    };
    _Float16* W1h  = (_Float16*)alloc((size_t)128 * 576  * 2);
    _Float16* W2h  = (_Float16*)alloc((size_t)128 * 1152 * 2);
    _Float16* FW1h = (_Float16*)alloc((size_t)256 * 6272 * 2);
    _Float16* X2   = (_Float16*)alloc((size_t)NROI * FIN * 2);   // 51.4 MB
    float*    FC1  = (float*)   alloc((size_t)NROI * FC_DIM * 4);

    pack_weights   <<<6272, 256, 0, stream>>>(w1, w2, fw1, W1h, W2h, FW1h);
    roi_conv_kernel<<<NROI, 128, 0, stream>>>(feat, boxes, W1h, b1, W2h, b2, X2);
    fc1_kernel     <<<256, 128, 0, stream>>>(X2, FW1h, fb1, FC1);
    fc2_kernel     <<<NROI / 4, 128, 0, stream>>>(FC1, fw2, fb2, out);
}
